// GAT_51445118271861
// MI455X (gfx1250) — matrix operand order repaired and checked by reading, not compiled, not run on hardware
//
#include <hip/hip_runtime.h>
#include <hip/hip_bf16.h>
#include <stdint.h>

// ---------------- problem constants (match reference) ----------------
constexpr int N_NODES = 50000;
constexpr int E_EDGES = 800000;
constexpr int IN_C    = 128;
constexpr int HID     = 32;
constexpr int HEADS   = 8;
constexpr int F1      = HEADS * HID;   // 256
constexpr int OUT_C   = 64;
constexpr float NEG_SLOPE = 0.2f;

typedef unsigned short ushort_t;

// ---------------- WMMA types ----------------
typedef __attribute__((ext_vector_type(16))) __bf16 v16bf;
typedef __attribute__((ext_vector_type(8)))  float  v8f;

union Frag { v16bf v; uint4 q[2]; };   // 32 bytes: 16 bf16 = 2 x b128 loads

__device__ __forceinline__ ushort_t f2bf_bits(float f) {
  // round-to-nearest-even f32 -> bf16 bit pattern
  unsigned u = __float_as_uint(f);
  u += 0x7FFFu + ((u >> 16) & 1u);
  return (ushort_t)(u >> 16);
}

// monotonic uint encoding of float for atomicMax-based segment max
__device__ __forceinline__ unsigned fenc(float x) {
  unsigned u = __float_as_uint(x);
  return (u & 0x80000000u) ? ~u : (u | 0x80000000u);
}
__device__ __forceinline__ float fdec(unsigned k) {
  unsigned u = (k & 0x80000000u) ? (k & 0x7FFFFFFFu) : ~k;
  return __uint_as_float(u);
}

// ---------------- zero fill (b128 stores) ----------------
__global__ void fill_zero_v4(uint4* __restrict__ p, size_t n4) {
  size_t i = (size_t)blockIdx.x * blockDim.x + threadIdx.x;
  size_t stride = (size_t)gridDim.x * blockDim.x;
  uint4 z = {0u, 0u, 0u, 0u};
  for (; i < n4; i += stride) p[i] = z;
}

// ---------------- f32 -> bf16 convert (2-wide packed stores) ----------------
__global__ void conv_bf16_kernel(const float2* __restrict__ in,
                                 unsigned* __restrict__ out, size_t npairs) {
  size_t i = (size_t)blockIdx.x * blockDim.x + threadIdx.x;
  size_t stride = (size_t)gridDim.x * blockDim.x;
  for (; i < npairs; i += stride) {
    float2 v = in[i];
    out[i] = (unsigned)f2bf_bits(v.x) | ((unsigned)f2bf_bits(v.y) << 16);
  }
}

// ---------------- pack W[K x NCOLS] (f32, row-major) into bf16 B-fragment order
// out[(colTile*(K/32)+ks)*512 + lane*16 + j] = bf16(W[ks*32 + 16*(lane>>4) + j][colTile*16 + (lane&15)])
__global__ void pack_w_bf16(const float* __restrict__ W, ushort_t* __restrict__ out,
                            int K, int NCOLS) {
  int idx = blockIdx.x * blockDim.x + threadIdx.x;
  if (idx >= K * NCOLS) return;
  int t = idx >> 9;            // /512
  int r = idx & 511;
  int lane = r >> 4, j = r & 15;
  int ksteps = K >> 5;
  int colTile = t / ksteps, ks = t - colTile * ksteps;
  int k = ks * 32 + ((lane >> 4) << 4) + j;
  int n = colTile * 16 + (lane & 15);
  out[idx] = f2bf_bits(W[(size_t)k * NCOLS + n]);
}

// ---------------- WMMA GEMM: Y[Nrows x NCOLS] = Xbf[Nrows x K] @ Wpk
// 4 waves / block, one 16x16 output tile per wave, all loads b128.
template <int K, int NCOLS>
__global__ __launch_bounds__(128)
void gemm_wmma_bf16pk(const ushort_t* __restrict__ Xbf,
                      const ushort_t* __restrict__ Wpk,
                      float* __restrict__ Y) {
  constexpr int KSTEPS = K / 32;
  const int row0 = blockIdx.x * 16;
  const int wave = threadIdx.x >> 5;
  const int lane = threadIdx.x & 31;
  const int colTile = blockIdx.y * 4 + wave;
  const int m  = lane & 15;
  const int kh = lane >> 4;
  // A: row (row0+m), halves [16*kh + 0..15] per 32-k step
  const ushort_t* arow = Xbf + (size_t)(row0 + m) * K + (kh << 4);
  // B: pre-packed fragment order, 16 contiguous halves per lane per k-step
  const ushort_t* bptr = Wpk + (size_t)colTile * KSTEPS * 512 + lane * 16;
  v8f acc = {};
#pragma unroll
  for (int ks = 0; ks < KSTEPS; ++ks) {
    Frag a, b;
    a.q[0] = *(const uint4*)(arow + ks * 32);
    a.q[1] = *(const uint4*)(arow + ks * 32 + 8);
    b.q[0] = *(const uint4*)(bptr + ks * 512);
    b.q[1] = *(const uint4*)(bptr + ks * 512 + 8);
    acc = __builtin_amdgcn_wmma_f32_16x16x32_bf16(
        /*neg_a=*/false, a.v, /*neg_b=*/false, b.v,
        /*c_mod=*/(short)0, acc, /*reuse_a=*/false, /*reuse_b=*/false);
  }
#pragma unroll
  for (int r = 0; r < 8; ++r) {   // C/D layout: M = r + 8*kh, N = lane&15
    Y[(size_t)(row0 + r + (kh << 3)) * NCOLS + colTile * 16 + m] = acc[r];
  }
}

// ---------------- layer-1 alpha dots: wave per (node, head), lane = channel
__global__ __launch_bounds__(256)
void alpha1_kernel(const float* __restrict__ h, const float* __restrict__ a_src,
                   const float* __restrict__ a_dst, float* __restrict__ as,
                   float* __restrict__ ad) {
  const int n = blockIdx.x;
  const int head = threadIdx.x >> 5;
  const int c    = threadIdx.x & 31;
  float hv = h[(size_t)n * F1 + head * HID + c];
  float s = hv * a_src[head * HID + c];
  float d = hv * a_dst[head * HID + c];
#pragma unroll
  for (int off = 16; off > 0; off >>= 1) {
    s += __shfl_down(s, off, 32);
    d += __shfl_down(d, off, 32);
  }
  if (c == 0) { as[(size_t)n * HEADS + head] = s; ad[(size_t)n * HEADS + head] = d; }
}

// ---------------- layer-2 alpha dots: wave per node, 64 channels
__global__ __launch_bounds__(256)
void alpha2_kernel(const float* __restrict__ g, const float* __restrict__ a_src,
                   const float* __restrict__ a_dst, float* __restrict__ as,
                   float* __restrict__ ad, int n_nodes) {
  const int n = blockIdx.x * (blockDim.x >> 5) + (threadIdx.x >> 5);
  const int lane = threadIdx.x & 31;
  if (n >= n_nodes) return;
  float s = 0.f, d = 0.f;
#pragma unroll
  for (int c = lane; c < OUT_C; c += 32) {
    float gv = g[(size_t)n * OUT_C + c];
    s += gv * a_src[c];
    d += gv * a_dst[c];
  }
#pragma unroll
  for (int off = 16; off > 0; off >>= 1) {
    s += __shfl_down(s, off, 32);
    d += __shfl_down(d, off, 32);
  }
  if (lane == 0) { as[n] = s; ad[n] = d; }
}

// ---------------- edge pass 1: segment max (encoded-uint atomicMax)
template <int H>
__global__ void edge_max_kernel(const long long* __restrict__ src,
                                const long long* __restrict__ dst,
                                const float* __restrict__ as,
                                const float* __restrict__ ad,
                                unsigned* __restrict__ mkey, int total) {
  int idx = blockIdx.x * blockDim.x + threadIdx.x;
  if (idx >= total) return;
  int e = idx / H, h = idx - e * H;
  long long s = src[e], d = dst[e];
  float v = as[(size_t)s * H + h] + ad[(size_t)d * H + h];
  v = (v >= 0.f) ? v : NEG_SLOPE * v;
  atomicMax(&mkey[(size_t)d * H + h], fenc(v));
}

// ---------------- edge pass 2: w = exp(e - m[dst]); z[dst] += w
template <int H>
__global__ void edge_expsum_kernel(const long long* __restrict__ src,
                                   const long long* __restrict__ dst,
                                   const float* __restrict__ as,
                                   const float* __restrict__ ad,
                                   const unsigned* __restrict__ mkey,
                                   float* __restrict__ w, float* __restrict__ z,
                                   int total) {
  int idx = blockIdx.x * blockDim.x + threadIdx.x;
  if (idx >= total) return;
  int e = idx / H, h = idx - e * H;
  long long s = src[e], d = dst[e];
  float v = as[(size_t)s * H + h] + ad[(size_t)d * H + h];
  v = (v >= 0.f) ? v : NEG_SLOPE * v;
  float ww = expf(v - fdec(mkey[(size_t)d * H + h]));
  w[(size_t)e * H + h] = ww;
  atomicAdd(&z[(size_t)d * H + h], ww);
}

// ---------------- layer-1 scatter: block(256) per edge covers all (head,chan)
__global__ __launch_bounds__(256)
void edge_scatter1_kernel(const long long* __restrict__ src,
                          const long long* __restrict__ dst,
                          const float* __restrict__ h,
                          const float* __restrict__ w,
                          const float* __restrict__ z,
                          float* __restrict__ agg) {
  const int e = blockIdx.x;
  const int head = threadIdx.x >> 5;
  const int c    = threadIdx.x & 31;
  long long s = src[e], d = dst[e];
  float alpha = w[(size_t)e * HEADS + head] / z[(size_t)d * HEADS + head];
  atomicAdd(&agg[(size_t)d * F1 + head * HID + c],
            h[(size_t)s * F1 + head * HID + c] * alpha);
}

// ---------------- layer-2 scatter: thread per (edge, channel)
__global__ void edge_scatter2_kernel(const long long* __restrict__ src,
                                     const long long* __restrict__ dst,
                                     const float* __restrict__ g,
                                     const float* __restrict__ w,
                                     const float* __restrict__ z,
                                     float* __restrict__ out, int total) {
  int idx = blockIdx.x * blockDim.x + threadIdx.x;
  if (idx >= total) return;
  int e = idx >> 6, c = idx & 63;
  long long s = src[e], d = dst[e];
  float alpha = w[e] / z[d];
  atomicAdd(&out[(size_t)d * OUT_C + c], g[(size_t)s * OUT_C + c] * alpha);
}

// ---------------- layer-1 epilogue: h2(bf16) = ELU(agg + b1), 2-wide packed
__global__ void elu_bias_bf16_kernel(const float2* __restrict__ agg,
                                     const float* __restrict__ b,
                                     unsigned* __restrict__ out, size_t npairs) {
  size_t i = (size_t)blockIdx.x * blockDim.x + threadIdx.x;
  size_t stride = (size_t)gridDim.x * blockDim.x;
  for (; i < npairs; i += stride) {
    float2 v = agg[i];
    int cb = (int)((2 * i) % F1);
    float v0 = v.x + b[cb];
    float v1 = v.y + b[cb + 1];
    v0 = (v0 > 0.f) ? v0 : (expf(v0) - 1.f);
    v1 = (v1 > 0.f) ? v1 : (expf(v1) - 1.f);
    out[i] = (unsigned)f2bf_bits(v0) | ((unsigned)f2bf_bits(v1) << 16);
  }
}

// ---------------- final bias (in-place on d_out)
__global__ void bias2_kernel(float* __restrict__ out, const float* __restrict__ b,
                             size_t total) {
  size_t i = (size_t)blockIdx.x * blockDim.x + threadIdx.x;
  size_t stride = (size_t)gridDim.x * blockDim.x;
  for (; i < total; i += stride) out[i] += b[i % OUT_C];
}

extern "C" void kernel_launch(void* const* d_in, const int* in_sizes, int n_in,
                              void* d_out, int out_size, void* d_ws, size_t ws_size,
                              hipStream_t stream) {
  const float*     x      = (const float*)d_in[0];
  const long long* eidx   = (const long long*)d_in[1];   // int64 [2, E]
  const float*     W1     = (const float*)d_in[2];
  const float*     a1_src = (const float*)d_in[3];
  const float*     a1_dst = (const float*)d_in[4];
  const float*     b1     = (const float*)d_in[5];
  const float*     W2     = (const float*)d_in[6];
  const float*     a2_src = (const float*)d_in[7];
  const float*     a2_dst = (const float*)d_in[8];
  const float*     b2     = (const float*)d_in[9];
  const long long* src = eidx;
  const long long* dst = eidx + E_EDGES;
  float* out = (float*)d_out;

  // -------- workspace layout (floats; every block multiple of 4 => 16B aligned)
  float* ws = (float*)d_ws;
  size_t off = 0;
  float*    h1   = ws + off; off += (size_t)N_NODES * F1;    // 12.8M (later: h2 bf16)
  float*    agg1 = ws + off; off += (size_t)N_NODES * F1;    // 12.8M (later: g f32)
  float*    we1  = ws + off; off += (size_t)E_EDGES * HEADS; // 6.4M  (early: xbf; later: we2)
  float*    as1  = ws + off; off += (size_t)N_NODES * HEADS;
  float*    ad1  = ws + off; off += (size_t)N_NODES * HEADS;
  unsigned* m1   = (unsigned*)(ws + off); off += (size_t)N_NODES * HEADS;
  float*    z1   = ws + off; off += (size_t)N_NODES * HEADS;
  float*    as2  = ws + off; off += (size_t)N_NODES;
  float*    ad2  = ws + off; off += (size_t)N_NODES;
  unsigned* m2   = (unsigned*)(ws + off); off += (size_t)N_NODES;
  float*    z2   = ws + off; off += (size_t)N_NODES;
  ushort_t* wpk1 = (ushort_t*)(ws + off); off += (size_t)(IN_C * F1) / 2;   // 32768 bf16
  ushort_t* wpk2 = (ushort_t*)(ws + off); off += (size_t)(F1 * OUT_C) / 2;  // 16384 bf16
  ushort_t* xbf  = (ushort_t*)we1;   // N*IN_C bf16 (12.8MB) inside we1 (25.6MB), dead before we1 use
  ushort_t* h2bf = (ushort_t*)h1;    // N*F1 bf16 (25.6MB) inside h1 (51.2MB), h1 dead after scatter1
  float*    g    = agg1;             // N*OUT_C f32, agg1 dead after ELU
  float*    we2  = we1;

  const int TB = 256;
  auto blocks = [](size_t n, int tb) { return (unsigned)((n + tb - 1) / tb); };

  // ---- zero init: agg1, m1+z1 (contiguous), m2+z2 (contiguous), d_out
  fill_zero_v4<<<blocks((size_t)N_NODES * F1 / 4, TB), TB, 0, stream>>>(
      (uint4*)agg1, (size_t)N_NODES * F1 / 4);
  fill_zero_v4<<<blocks((size_t)N_NODES * HEADS * 2 / 4, TB), TB, 0, stream>>>(
      (uint4*)m1, (size_t)N_NODES * HEADS * 2 / 4);
  fill_zero_v4<<<blocks((size_t)N_NODES * 2 / 4, TB), TB, 0, stream>>>(
      (uint4*)m2, (size_t)N_NODES * 2 / 4);
  fill_zero_v4<<<blocks((size_t)N_NODES * OUT_C / 4, TB), TB, 0, stream>>>(
      (uint4*)out, (size_t)N_NODES * OUT_C / 4);

  // ---- operand prep: x -> bf16, pack W1/W2 into B-fragment order
  conv_bf16_kernel<<<2048, TB, 0, stream>>>((const float2*)x, (unsigned*)xbf,
                                            (size_t)N_NODES * IN_C / 2);
  pack_w_bf16<<<blocks(IN_C * F1, TB), TB, 0, stream>>>(W1, wpk1, IN_C, F1);
  pack_w_bf16<<<blocks(F1 * OUT_C, TB), TB, 0, stream>>>(W2, wpk2, F1, OUT_C);

  // ---- layer 1 ----
  gemm_wmma_bf16pk<IN_C, F1><<<dim3(N_NODES / 16, F1 / 16 / 4), 128, 0, stream>>>(
      xbf, wpk1, h1);
  alpha1_kernel<<<N_NODES, 256, 0, stream>>>(h1, a1_src, a1_dst, as1, ad1);
  {
    int total = E_EDGES * HEADS;
    edge_max_kernel<HEADS><<<blocks(total, TB), TB, 0, stream>>>(src, dst, as1, ad1,
                                                                 m1, total);
    edge_expsum_kernel<HEADS><<<blocks(total, TB), TB, 0, stream>>>(
        src, dst, as1, ad1, m1, we1, z1, total);
  }
  edge_scatter1_kernel<<<E_EDGES, 256, 0, stream>>>(src, dst, h1, we1, z1, agg1);
  elu_bias_bf16_kernel<<<4096, TB, 0, stream>>>((const float2*)agg1, b1,
                                                (unsigned*)h2bf,
                                                (size_t)N_NODES * F1 / 2);

  // ---- layer 2 ----
  gemm_wmma_bf16pk<F1, OUT_C><<<dim3(N_NODES / 16, 1), 128, 0, stream>>>(
      h2bf, wpk2, g);
  alpha2_kernel<<<blocks(N_NODES, 8), 256, 0, stream>>>(g, a2_src, a2_dst, as2, ad2,
                                                        N_NODES);
  {
    int total = E_EDGES;
    edge_max_kernel<1><<<blocks(total, TB), TB, 0, stream>>>(src, dst, as2, ad2, m2,
                                                             total);
    edge_expsum_kernel<1><<<blocks(total, TB), TB, 0, stream>>>(src, dst, as2, ad2,
                                                                m2, we2, z2, total);
  }
  {
    int total = E_EDGES * OUT_C;
    edge_scatter2_kernel<<<blocks(total, TB), TB, 0, stream>>>(src, dst, g, we2, z2,
                                                               out, total);
  }
  bias2_kernel<<<2048, TB, 0, stream>>>(out, b2, (size_t)N_NODES * OUT_C);
}
